// F2NetHead_11931419148456
// MI455X (gfx1250) — matrix operand order repaired and checked by reading, not compiled, not run on hardware
//
#include <hip/hip_runtime.h>
#include <math.h>

// ---------------------------------------------------------------------------
// F2NetHead for MI455X (gfx1250): bf16 WMMA pipeline with async-LDS staging.
// D = 1024, S = 4096, B = 4, N = 16384. All GEMMs are [N,1024]x[1024,1024]^T.
// ---------------------------------------------------------------------------

#define DM 1024
#define SEQ 4096

typedef __attribute__((ext_vector_type(16))) __bf16 v16bf;
typedef __attribute__((ext_vector_type(8)))  float  v8f;

union Frag {
    v16bf v;
    uint4 u[2];
};

__device__ __forceinline__ unsigned short f2bf(float f) {
    unsigned int u = __builtin_bit_cast(unsigned int, f);
    u += 0x7FFFu + ((u >> 16) & 1u);   // round-to-nearest-even
    return (unsigned short)(u >> 16);
}

__device__ __forceinline__ float silu_f(float v) {
    return v / (1.0f + expf(-v));
}
__device__ __forceinline__ float gelu_f(float v) {
    return 0.5f * v * (1.0f + erff(v * 0.70710678118654752f));
}

// gfx1250 async copy: global -> LDS, 128 bits per lane, tracked by ASYNCcnt.
__device__ __forceinline__ void async_b128(const unsigned short* g,
                                           const unsigned short* l) {
    unsigned lds = (unsigned)(unsigned long long)(uintptr_t)l;   // LDS = addr[31:0]
    unsigned long long ga = (unsigned long long)(uintptr_t)g;
    asm volatile("global_load_async_to_lds_b128 %0, %1, off"
                 :: "v"(lds), "v"(ga) : "memory");
}

#if __has_builtin(__builtin_amdgcn_s_wait_asynccnt)
#define WAIT_ASYNC(n) __builtin_amdgcn_s_wait_asynccnt(n)
#else
#define WAIT_ASYNC(n) asm volatile("s_wait_asynccnt %0" :: "i"(n) : "memory")
#endif

// ---------------------------------------------------------------------------
// fp32 -> bf16 conversion (grid-stride)
// ---------------------------------------------------------------------------
__global__ void cvt_bf16_kernel(const float* __restrict__ src,
                                unsigned short* __restrict__ dst, int n) {
    for (int i = blockIdx.x * blockDim.x + threadIdx.x; i < n;
         i += gridDim.x * blockDim.x)
        dst[i] = f2bf(src[i]);
}

// conv_w [o][i][k] (o*3072 + i*3 + k) -> planes [k][o][i] in bf16
__global__ void conv_repack_kernel(const float* __restrict__ cw,
                                   unsigned short* __restrict__ dst) {
    const int n = 3 * DM * DM;
    for (int idx = blockIdx.x * blockDim.x + threadIdx.x; idx < n;
         idx += gridDim.x * blockDim.x) {
        int k = idx / (DM * DM);
        int rem = idx - k * (DM * DM);
        int o = rem >> 10;
        int i = rem & (DM - 1);
        dst[(size_t)k * DM * DM + (size_t)o * DM + i] =
            f2bf(cw[(size_t)o * (DM * 3) + i * 3 + k]);
    }
}

// ---------------------------------------------------------------------------
// Generic bf16 GEMM:  C[m][n] = sum_k A[m][k] * W[n][k]   (einsum bsd,ed->bse)
// Block = 256 thr (8 waves), block tile 64(M) x 128(N); wave tile 32x32,
// 2x2 fragments of v_wmma_f32_16x16x32_bf16, fp32 accumulate.
// A (64x32) and B (128x32) tiles double-buffered in LDS, filled via
// global_load_async_to_lds_b128 (3 async ops per thread per k-step),
// consumed via ds_load_b128.
// ---------------------------------------------------------------------------
__global__ __launch_bounds__(256)
void gemm_bf16_kernel(const unsigned short* __restrict__ A,
                      const unsigned short* __restrict__ W,
                      const float* __restrict__ bias, float scale,
                      float* __restrict__ outF,
                      unsigned short* __restrict__ outH) {
    __shared__ __align__(16) unsigned short shA[2][64 * 32];
    __shared__ __align__(16) unsigned short shB[2][128 * 32];

    const int tid  = threadIdx.x;
    const int lane = tid & 31;
    const int wv   = tid >> 5;
    const int wm   = wv >> 2;       // 0..1
    const int wn   = wv & 3;        // 0..3
    const int ml   = lane & 15;
    const int hh   = lane >> 4;

    // staging assignment: one 16B segment per thread for A, two for B
    const int rowT = tid >> 2;      // 0..63
    const int segT = (tid & 3) * 8; // half offset within 32-wide k slab
    const unsigned short* gA  = A + (size_t)(blockIdx.x * 64 + rowT) * DM + segT;
    const unsigned short* gB0 = W + (size_t)(blockIdx.y * 128 + rowT) * DM + segT;
    const unsigned short* gB1 = W + (size_t)(blockIdx.y * 128 + 64 + rowT) * DM + segT;
    const int aOff  = rowT * 32 + segT;
    const int bOff0 = rowT * 32 + segT;
    const int bOff1 = (64 + rowT) * 32 + segT;

    v8f acc[2][2] = {};

    // fragment read offsets (halfs) within a staged tile
    const int ra0 = (wm * 32 + ml) * 32;
    const int ra1 = (wm * 32 + 16 + ml) * 32;
    const int cb0 = (wn * 32 + ml) * 32;
    const int cb1 = (wn * 32 + 16 + ml) * 32;

    // prologue: stage 0 into buffer 0
    async_b128(gA, &shA[0][aOff]);
    async_b128(gB0, &shB[0][bOff0]);
    async_b128(gB1, &shB[0][bOff1]);

    for (int t = 0; t < 32; ++t) {
        const int buf = t & 1;
        if (t < 31) {
            const int kc = (t + 1) * 32;
            const int nb = buf ^ 1;
            async_b128(gA + kc, &shA[nb][aOff]);
            async_b128(gB0 + kc, &shB[nb][bOff0]);
            async_b128(gB1 + kc, &shB[nb][bOff1]);
            WAIT_ASYNC(3);          // stage t landed; stage t+1 in flight
        } else {
            WAIT_ASYNC(0);
        }
        __syncthreads();

        const unsigned short* sA = &shA[buf][0];
        const unsigned short* sB = &shB[buf][0];
        Frag a0, a1, b0, b1;
        // A-fragment (16-bit A 16x32): lane(m,h) holds K = 8h..8h+7 and
        // 16+8h..16+8h+7
        a0.u[0] = *(const uint4*)(sA + ra0 + 8 * hh);
        a0.u[1] = *(const uint4*)(sA + ra0 + 16 + 8 * hh);
        a1.u[0] = *(const uint4*)(sA + ra1 + 8 * hh);
        a1.u[1] = *(const uint4*)(sA + ra1 + 16 + 8 * hh);
        // B-fragment (32x16): lane(n,h) holds K = 8h..8h+7 and 16+8h..16+8h+7 of column n
        b0.u[0] = *(const uint4*)(sB + cb0 + 8 * hh);
        b0.u[1] = *(const uint4*)(sB + cb0 + 16 + 8 * hh);
        b1.u[0] = *(const uint4*)(sB + cb1 + 8 * hh);
        b1.u[1] = *(const uint4*)(sB + cb1 + 16 + 8 * hh);

        acc[0][0] = __builtin_amdgcn_wmma_f32_16x16x32_bf16(
            false, a0.v, false, b0.v, (short)0, acc[0][0], false, false);
        acc[0][1] = __builtin_amdgcn_wmma_f32_16x16x32_bf16(
            false, a0.v, false, b1.v, (short)0, acc[0][1], false, false);
        acc[1][0] = __builtin_amdgcn_wmma_f32_16x16x32_bf16(
            false, a1.v, false, b0.v, (short)0, acc[1][0], false, false);
        acc[1][1] = __builtin_amdgcn_wmma_f32_16x16x32_bf16(
            false, a1.v, false, b1.v, (short)0, acc[1][1], false, false);

        __syncthreads();            // protect buf before it is re-staged
    }

    const int mr = blockIdx.x * 64 + wm * 32;
    const int nc = blockIdx.y * 128 + wn * 32;
    // C/D layout: VGPR r holds M = r (lanes 0-15) / M = 8+r (lanes 16-31)
    for (int i = 0; i < 2; ++i)
        for (int j = 0; j < 2; ++j) {
            const int col = nc + 16 * j + ml;
            const float b = bias ? bias[col] : 0.0f;
            for (int r = 0; r < 8; ++r) {
                const int row = mr + 16 * i + r + 8 * hh;
                const float v = acc[i][j][r] * scale + b;
                const size_t o = (size_t)row * DM + col;
                if (outF) outF[o] = v;
                if (outH) outH[o] = f2bf(v);
            }
        }
}

// ---------------------------------------------------------------------------
// Conv-as-GEMM: ct[r][o] = sum_k sum_i Q[r+k-1][i] * convW[k][o][i] + cb[o]
// with zero padding at per-batch sequence edges (r mod 4096).
// Epilogue: H = gelu(silu(ct)) stored as bf16 (input to the wa GEMM).
// Direct global loads (edge masking is per-lane).
// ---------------------------------------------------------------------------
__global__ __launch_bounds__(256)
void convgemm_bf16_kernel(const unsigned short* __restrict__ Q,
                          const unsigned short* __restrict__ W3,
                          const float* __restrict__ cbias,
                          unsigned short* __restrict__ outH) {
    const int lane = threadIdx.x & 31;
    const int wv   = threadIdx.x >> 5;
    const int wm   = wv >> 2;
    const int wn   = wv & 3;
    const int mr   = blockIdx.x * 64 + wm * 32;
    const int nc   = blockIdx.y * 128 + wn * 32;
    const int ml   = lane & 15;
    const int hh   = lane >> 4;

    v8f acc[2][2] = {};
    const int row0 = mr + ml;
    const int row1 = mr + 16 + ml;

    for (int k = 0; k < 3; ++k) {
        const int shift = k - 1;
        const int sb0 = row0 & (SEQ - 1);
        const int sb1 = row1 & (SEQ - 1);
        const bool v0 = (shift == 0) || (shift < 0 ? (sb0 != 0) : (sb0 != SEQ - 1));
        const bool v1 = (shift == 0) || (shift < 0 ? (sb1 != 0) : (sb1 != SEQ - 1));
        const unsigned short* A0 = Q + (size_t)(row0 + shift) * DM;
        const unsigned short* A1 = Q + (size_t)(row1 + shift) * DM;
        const unsigned short* Wk = W3 + (size_t)k * DM * DM;
        const unsigned short* W0 = Wk + (size_t)(nc + ml) * DM;
        const unsigned short* W1 = Wk + (size_t)(nc + 16 + ml) * DM;
        const uint4 zz = {0u, 0u, 0u, 0u};

        for (int kc = 0; kc < DM; kc += 32) {
            Frag a0, a1, b0, b1;
            a0.u[0] = v0 ? *(const uint4*)(A0 + kc + 8 * hh) : zz;
            a0.u[1] = v0 ? *(const uint4*)(A0 + kc + 16 + 8 * hh) : zz;
            a1.u[0] = v1 ? *(const uint4*)(A1 + kc + 8 * hh) : zz;
            a1.u[1] = v1 ? *(const uint4*)(A1 + kc + 16 + 8 * hh) : zz;
            b0.u[0] = *(const uint4*)(W0 + kc + 8 * hh);
            b0.u[1] = *(const uint4*)(W0 + kc + 16 + 8 * hh);
            b1.u[0] = *(const uint4*)(W1 + kc + 8 * hh);
            b1.u[1] = *(const uint4*)(W1 + kc + 16 + 8 * hh);

            acc[0][0] = __builtin_amdgcn_wmma_f32_16x16x32_bf16(
                false, a0.v, false, b0.v, (short)0, acc[0][0], false, false);
            acc[0][1] = __builtin_amdgcn_wmma_f32_16x16x32_bf16(
                false, a0.v, false, b1.v, (short)0, acc[0][1], false, false);
            acc[1][0] = __builtin_amdgcn_wmma_f32_16x16x32_bf16(
                false, a1.v, false, b0.v, (short)0, acc[1][0], false, false);
            acc[1][1] = __builtin_amdgcn_wmma_f32_16x16x32_bf16(
                false, a1.v, false, b1.v, (short)0, acc[1][1], false, false);
        }
    }

    for (int i = 0; i < 2; ++i)
        for (int j = 0; j < 2; ++j) {
            const int col = nc + 16 * j + ml;
            const float b = cbias[col];
            for (int r = 0; r < 8; ++r) {
                const int row = mr + 16 * i + r + 8 * hh;
                const float ct = acc[i][j][r] + b;
                outH[(size_t)row * DM + col] = f2bf(gelu_f(silu_f(ct)));
            }
        }
}

// ---------------------------------------------------------------------------
// Column softmax over sequence + weighted sum of V:
//   a[b,s,e] = softmax_s(Z[b,s,e]);  out[b,e] = sum_s V[b,s,e] * a[b,s,e]
// Block: 256 thr = 16 columns x 16 row-groups; grid (DM/16, B).
// ---------------------------------------------------------------------------
__global__ __launch_bounds__(256)
void colsoftmax_wsum_kernel(const float* __restrict__ Z,
                            const float* __restrict__ V,
                            float* __restrict__ out) {
    __shared__ float sm[256], sl[256];
    __shared__ float cm[16], cl[16];
    const int b  = blockIdx.y;
    const int e0 = blockIdx.x * 16;
    const int c  = threadIdx.x & 15;
    const int rg = threadIdx.x >> 4;
    const size_t base = ((size_t)b * SEQ) * DM + e0 + c;

    float m = -3.4e38f, l = 0.0f;
    for (int s = rg; s < SEQ; s += 16) {
        const float z = Z[base + (size_t)s * DM];
        const float nm = fmaxf(m, z);
        l = l * expf(m - nm) + expf(z - nm);
        m = nm;
    }
    sm[threadIdx.x] = m;
    sl[threadIdx.x] = l;
    __syncthreads();
    if (rg == 0) {
        float M = -3.4e38f;
        for (int t = 0; t < 16; ++t) M = fmaxf(M, sm[t * 16 + c]);
        float L = 0.0f;
        for (int t = 0; t < 16; ++t) L += sl[t * 16 + c] * expf(sm[t * 16 + c] - M);
        cm[c] = M;
        cl[c] = L;
    }
    __syncthreads();
    const float M = cm[c], L = cl[c];
    float acc = 0.0f;
    for (int s = rg; s < SEQ; s += 16) {
        const size_t idx = base + (size_t)s * DM;
        acc += V[idx] * expf(Z[idx] - M);
    }
    __syncthreads();
    sm[threadIdx.x] = acc;
    __syncthreads();
    if (rg == 0) {
        float A = 0.0f;
        for (int t = 0; t < 16; ++t) A += sm[t * 16 + c];
        out[(size_t)b * DM + e0 + c] = A / L;
    }
}

// P = O * s1[b,e] (in place over O); Pb = bf16(gelu(P))
__global__ void ew_P_kernel(float* __restrict__ OP,
                            const float* __restrict__ s1,
                            unsigned short* __restrict__ Pb, int n) {
    for (int idx = blockIdx.x * blockDim.x + threadIdx.x; idx < n;
         idx += gridDim.x * blockDim.x) {
        const int row = idx >> 10;
        const int bb  = row >> 12;          // row / 4096
        const int e   = idx & (DM - 1);
        const float p = OP[idx] * s1[bb * DM + e];
        OP[idx] = p;
        Pb[idx] = f2bf(gelu_f(p));
    }
}

// Lb = bf16( silu(G) * s2[b,e] )
__global__ void ew_L_kernel(const float* __restrict__ G,
                            const float* __restrict__ s2,
                            unsigned short* __restrict__ Lb, int n) {
    for (int idx = blockIdx.x * blockDim.x + threadIdx.x; idx < n;
         idx += gridDim.x * blockDim.x) {
        const int row = idx >> 10;
        const int bb  = row >> 12;
        const int e   = idx & (DM - 1);
        Lb[idx] = f2bf(silu_f(G[idx]) * s2[bb * DM + e]);
    }
}

// ---------------------------------------------------------------------------
extern "C" void kernel_launch(void* const* d_in, const int* in_sizes, int n_in,
                              void* d_out, int out_size, void* d_ws, size_t ws_size,
                              hipStream_t stream) {
    const float* x      = (const float*)d_in[0];
    const float* Wq_w   = (const float*)d_in[1];
    const float* Wq_b   = (const float*)d_in[2];
    const float* Wo_w   = (const float*)d_in[3];
    const float* Wo_b   = (const float*)d_in[4];
    const float* Wg_w   = (const float*)d_in[5];
    const float* Wg_b   = (const float*)d_in[6];
    const float* wa_w   = (const float*)d_in[7];
    const float* wb_w   = (const float*)d_in[8];
    const float* Wout_w = (const float*)d_in[9];
    const float* Wout_b = (const float*)d_in[10];
    const float* conv_w = (const float*)d_in[11];
    const float* conv_b = (const float*)d_in[12];

    const int M = in_sizes[0] / DM;          // 16384 rows
    const int B = M / SEQ;                   // 4 batches
    const int NE = M * DM;                   // elementwise count

    // workspace carve-up (256-byte aligned)
    char* p = (char*)d_ws;
    auto alloc = [&](size_t bytes) -> char* {
        char* r = p;
        p += (bytes + 255) & ~(size_t)255;
        return r;
    };
    unsigned short* xb    = (unsigned short*)alloc((size_t)NE * 2);
    unsigned short* Wqb   = (unsigned short*)alloc((size_t)DM * DM * 2);
    unsigned short* Wob   = (unsigned short*)alloc((size_t)DM * DM * 2);
    unsigned short* Wgb   = (unsigned short*)alloc((size_t)DM * DM * 2);
    unsigned short* wab   = (unsigned short*)alloc((size_t)DM * DM * 2);
    unsigned short* wbb   = (unsigned short*)alloc((size_t)DM * DM * 2);
    unsigned short* Woutb = (unsigned short*)alloc((size_t)DM * DM * 2);
    unsigned short* convWb= (unsigned short*)alloc((size_t)3 * DM * DM * 2);
    float*          Qf    = (float*)alloc((size_t)NE * 4);
    unsigned short* Qb    = (unsigned short*)alloc((size_t)NE * 2); // later: gelu(P) bf16
    float*          Of    = (float*)alloc((size_t)NE * 4);          // later: P (in place)
    float*          Gf    = (float*)alloc((size_t)NE * 4);
    unsigned short* Hb    = (unsigned short*)alloc((size_t)NE * 2); // later: Lb
    float*          Zf    = (float*)alloc((size_t)NE * 4);          // reused for Z1/Z2
    float*          s1    = (float*)alloc((size_t)B * DM * 4);
    float*          s2    = (float*)alloc((size_t)B * DM * 4);
    (void)ws_size; (void)n_in; (void)out_size;

    const float scale = 0.03125f;            // 1/sqrt(1024)
    const dim3 blk(256);
    const dim3 gemmGrid(M / 64, DM / 128);
    const int cvtBlocks = 4096;

    // stage 0: convert to bf16
    cvt_bf16_kernel<<<cvtBlocks, blk, 0, stream>>>(x, xb, NE);
    cvt_bf16_kernel<<<512, blk, 0, stream>>>(Wq_w, Wqb, DM * DM);
    cvt_bf16_kernel<<<512, blk, 0, stream>>>(Wo_w, Wob, DM * DM);
    cvt_bf16_kernel<<<512, blk, 0, stream>>>(Wg_w, Wgb, DM * DM);
    cvt_bf16_kernel<<<512, blk, 0, stream>>>(wa_w, wab, DM * DM);
    cvt_bf16_kernel<<<512, blk, 0, stream>>>(wb_w, wbb, DM * DM);
    cvt_bf16_kernel<<<512, blk, 0, stream>>>(Wout_w, Woutb, DM * DM);
    conv_repack_kernel<<<1536, blk, 0, stream>>>(conv_w, convWb);

    // stage 1: Q (fp32 + bf16), O, G
    gemm_bf16_kernel<<<gemmGrid, blk, 0, stream>>>(xb, Wqb, Wq_b, 1.0f, Qf, Qb);
    gemm_bf16_kernel<<<gemmGrid, blk, 0, stream>>>(xb, Wob, Wo_b, 1.0f, Of, nullptr);
    gemm_bf16_kernel<<<gemmGrid, blk, 0, stream>>>(xb, Wgb, Wg_b, 1.0f, Gf, nullptr);

    // stage 2: conv over sequence -> H = gelu(silu(ct)) bf16
    convgemm_bf16_kernel<<<gemmGrid, blk, 0, stream>>>(Qb, convWb, conv_b, Hb);

    // stage 3: Z1 = H @ wa^T * scale ; s1 = colsoftmax-weighted sum of Q
    gemm_bf16_kernel<<<gemmGrid, blk, 0, stream>>>(Hb, wab, nullptr, scale, Zf, nullptr);
    colsoftmax_wsum_kernel<<<dim3(DM / 16, B), blk, 0, stream>>>(Zf, Qf, s1);

    // stage 4: P = O * s1 (in place); Pb = bf16(gelu(P)) into Qb buffer
    ew_P_kernel<<<cvtBlocks, blk, 0, stream>>>(Of, s1, Qb, NE);

    // stage 5: Z2 = gelu(P) @ wb^T * scale ; s2 from P
    gemm_bf16_kernel<<<gemmGrid, blk, 0, stream>>>(Qb, wbb, nullptr, scale, Zf, nullptr);
    colsoftmax_wsum_kernel<<<dim3(DM / 16, B), blk, 0, stream>>>(Zf, Of, s2);

    // stage 6: Lb = bf16(silu(G) * s2) into Hb buffer
    ew_L_kernel<<<cvtBlocks, blk, 0, stream>>>(Gf, s2, Hb, NE);

    // stage 7: out = L @ Wout^T + b
    gemm_bf16_kernel<<<gemmGrid, blk, 0, stream>>>(Hb, Woutb, Wout_b, 1.0f,
                                                   (float*)d_out, nullptr);
}
